// DSQGBlock_35356170780877
// MI455X (gfx1250) — matrix operand order repaired and checked by reading, not compiled, not run on hardware
//
#include <hip/hip_runtime.h>
#include <hip/hip_bf16.h>

typedef __attribute__((ext_vector_type(16))) _Float16 v16h;
typedef __attribute__((ext_vector_type(8)))  float    v8f;

#define BM 128
#define BN 128
#define BK 32
#define LDSH 40   // halves per LDS row (32 data + 8 pad)

// ---------------- 44 causal offsets -----------------
__constant__ int c_off[44] = {
    0,1,2,3,4,5,6,7,8,9,10,11,12,13,14,15,16,17,18,19,20,21,22,23,24,25,26,27,
    28,29,30,31,32,48,64,96,128,192,256,384,512,768,1024,1536};

// -------- CDNA5 async memory->LDS copy (ASYNCcnt) ---
__device__ __forceinline__ void async_ld16(unsigned ldsAddr, const _Float16* gaddr) {
  asm volatile("global_load_async_to_lds_b128 %0, %1, off"
               :: "v"(ldsAddr), "v"((unsigned long long)(size_t)gaddr)
               : "memory");
}
#define WAIT_ASYNC_LE4() asm volatile("s_wait_asynccnt 0x4" ::: "memory")
#define WAIT_ASYNC_0()   asm volatile("s_wait_asynccnt 0x0" ::: "memory")

// ------- fp32[K][N] -> f16 transposed Wt[N][K] ------
// block (32,8), grid (N/32, K/32); LDS-tiled so both sides coalesce
__global__ __launch_bounds__(256) void cast_transpose_kernel(
    const float* __restrict__ W, _Float16* __restrict__ Wt, int K, int N) {
  __shared__ _Float16 tile[32][33];
  const int kb = blockIdx.y * 32, nb = blockIdx.x * 32;
#pragma unroll
  for (int r = 0; r < 32; r += 8) {
    int k = kb + threadIdx.y + r, n = nb + threadIdx.x;
    tile[threadIdx.y + r][threadIdx.x] = (_Float16)W[(size_t)k * N + n];
  }
  __syncthreads();
#pragma unroll
  for (int r = 0; r < 32; r += 8) {
    int n = nb + threadIdx.y + r, k = kb + threadIdx.x;
    Wt[(size_t)n * K + k] = tile[threadIdx.x][threadIdx.y + r];
  }
}

// ---------------- LayerNorm -> f16 ------------------
// one block (256 thr) per row of 1024
__global__ __launch_bounds__(256) void ln_kernel(
    const float* __restrict__ x, const float* __restrict__ g,
    const float* __restrict__ b, _Float16* __restrict__ out) {
  __shared__ float red[256];
  const int row = blockIdx.x, t = threadIdx.x;
  const float4* xp = (const float4*)(x + (size_t)row * 1024);
  float4 v = xp[t];
  red[t] = v.x + v.y + v.z + v.w;
  __syncthreads();
  for (int s = 128; s > 0; s >>= 1) { if (t < s) red[t] += red[t + s]; __syncthreads(); }
  float mu = red[0] * (1.0f / 1024.0f);
  __syncthreads();
  float dx = v.x - mu, dy = v.y - mu, dz = v.z - mu, dw = v.w - mu;
  red[t] = dx * dx + dy * dy + dz * dz + dw * dw;
  __syncthreads();
  for (int s = 128; s > 0; s >>= 1) { if (t < s) red[t] += red[t + s]; __syncthreads(); }
  float rstd = rsqrtf(red[0] * (1.0f / 1024.0f) + 1e-5f);
  const float4* gp = (const float4*)g;
  const float4* bp = (const float4*)b;
  float4 gv = gp[t], bv = bp[t];
  _Float16* op = out + (size_t)row * 1024 + t * 4;
  op[0] = (_Float16)(dx * rstd * gv.x + bv.x);
  op[1] = (_Float16)(dy * rstd * gv.y + bv.y);
  op[2] = (_Float16)(dz * rstd * gv.z + bv.z);
  op[3] = (_Float16)(dw * rstd * gv.w + bv.w);
}

// ---------------- WMMA f16 GEMM ---------------------
// C[M,N] = A[M,K](f16,row-major) @ Wt[N,K](f16, N-major i.e. B transposed) + bias
// MODE 0: outF = val   MODE 1: outF = val + res   MODE 2: outH = gelu(val)
// Double-buffered LDS fed by global_load_async_to_lds_b128 (ASYNCcnt pipeline).
template <int MODE>
__global__ __launch_bounds__(256) void gemm_wmma(
    const _Float16* __restrict__ A, const _Float16* __restrict__ Wt,
    const float* __restrict__ bias, const float* __restrict__ res,
    float* __restrict__ outF, _Float16* __restrict__ outH,
    int M, int N, int K) {
  __shared__ __align__(16) _Float16 sA[2][BM * LDSH];
  __shared__ __align__(16) _Float16 sB[2][BN * LDSH];  // sB[n][k]

  const int t = threadIdx.x;
  const int lane = t & 31, wave = t >> 5;
  const int wr = wave >> 1;   // 0..3 : 32 rows each
  const int wc = wave & 1;    // 0..1 : 64 cols each
  const int lh = lane & 15, hi = lane >> 4;
  const int m0 = blockIdx.y * BM, n0 = blockIdx.x * BN;

  v8f acc[2][4];
#pragma unroll
  for (int i = 0; i < 2; ++i)
#pragma unroll
    for (int j = 0; j < 4; ++j)
#pragma unroll
      for (int e = 0; e < 8; ++e) acc[i][j][e] = 0.0f;

  // staging: 2 threads per row, 16 halves (2x b128) each, for A and for B
  const int srow = t >> 1, spart = (t & 1) << 4;

  auto stage = [&](int buf, int k0) {
    const _Float16* ga = A + (size_t)(m0 + srow) * K + k0 + spart;
    unsigned la = (unsigned)(size_t)(&sA[buf][srow * LDSH + spart]);
    async_ld16(la,      ga);
    async_ld16(la + 16, ga + 8);
    const _Float16* gb = Wt + (size_t)(n0 + srow) * K + k0 + spart;
    unsigned lb = (unsigned)(size_t)(&sB[buf][srow * LDSH + spart]);
    async_ld16(lb,      gb);
    async_ld16(lb + 16, gb + 8);
  };

  const int T = K / BK;
  stage(0, 0);
  if (T > 1) stage(1, BK);

  for (int i = 0; i < T; ++i) {
    // async loads complete in order: <=4 outstanding => stage(i)'s 4 ops done
    if (i + 1 < T) { WAIT_ASYNC_LE4(); } else { WAIT_ASYNC_0(); }
    __syncthreads();

    const unsigned* A32 = (const unsigned*)(sA[i & 1]);
    const unsigned* B32 = (const unsigned*)(sB[i & 1]);

    // assemble fragments per ISA 7.12.2 (16-bit 16x32 layouts)
    v16h afr[2], bfr[4];
#pragma unroll
    for (int tm = 0; tm < 2; ++tm) {
      int m = wr * 32 + tm * 16 + lh;
      unsigned* au = (unsigned*)&afr[tm];
#pragma unroll
      for (int v = 0; v < 8; ++v) {
        int kk = 2 * (v & 3) + 16 * (v >> 2) + 8 * hi;  // K index of half-pair
        au[v] = A32[m * (LDSH / 2) + (kk >> 1)];
      }
    }
#pragma unroll
    for (int tn = 0; tn < 4; ++tn) {
      int n = wc * 64 + tn * 16 + lh;
      unsigned* bu = (unsigned*)&bfr[tn];
#pragma unroll
      for (int v = 0; v < 8; ++v) {
        int kk = 2 * (v & 3) + 16 * (v >> 2) + 8 * hi;  // K index of half-pair (same order as A)
        bu[v] = B32[n * (LDSH / 2) + (kk >> 1)];
      }
    }
#pragma unroll
    for (int tm = 0; tm < 2; ++tm)
#pragma unroll
      for (int tn = 0; tn < 4; ++tn)
        acc[tm][tn] = __builtin_amdgcn_wmma_f32_16x16x32_f16(
            false, afr[tm], false, bfr[tn], (short)0, acc[tm][tn], false, false);

    __syncthreads();                       // all waves done reading buf (i&1)
    if (i + 2 < T) stage(i & 1, (i + 2) * BK);
  }

  // epilogue: lane lh = column, rows hi*8 + r
#pragma unroll
  for (int tm = 0; tm < 2; ++tm) {
    int mbase = m0 + wr * 32 + tm * 16 + hi * 8;
#pragma unroll
    for (int tn = 0; tn < 4; ++tn) {
      int n = n0 + wc * 64 + tn * 16 + lh;
      float bv = bias[n];
#pragma unroll
      for (int r = 0; r < 8; ++r) {
        size_t idx = (size_t)(mbase + r) * N + n;
        float val = acc[tm][tn][r] + bv;
        if (MODE == 0) {
          outF[idx] = val;
        } else if (MODE == 1) {
          outF[idx] = val + res[idx];
        } else {
          float ge = 0.5f * val * (1.0f + erff(val * 0.70710678118654752f));
          outH[idx] = (_Float16)ge;
        }
      }
    }
  }
}

// ---------------- offset attention ------------------
// qkv layout: [N][3*1024] (q | k | v), head h at cols h*64..h*64+63
// one thread per (position i, head h); whole working set lives in 192MB L2
__global__ __launch_bounds__(256) void attn_kernel(
    const float* __restrict__ qkv, const float* __restrict__ pos_bias,
    float* __restrict__ out, int N) {
  int t = blockIdx.x * 256 + threadIdx.x;
  int i = t >> 4, h = t & 15;
  if (i >= N) return;
  const float4* qp = (const float4*)(qkv + (size_t)i * 3072 + h * 64);
  float4 q[16];
#pragma unroll
  for (int j = 0; j < 16; ++j) q[j] = qp[j];

  float s[44];
#pragma unroll
  for (int o = 0; o < 44; ++o) {
    int d = c_off[o];
    if (i >= d) {
      const float4* kp = (const float4*)(qkv + (size_t)(i - d) * 3072 + 1024 + h * 64);
      float dot = 0.0f;
#pragma unroll
      for (int j = 0; j < 16; ++j) {
        float4 kv = kp[j];
        dot += q[j].x * kv.x + q[j].y * kv.y + q[j].z * kv.z + q[j].w * kv.w;
      }
      s[o] = dot * 0.125f + pos_bias[o * 16 + h];
    } else {
      s[o] = -INFINITY;
    }
  }
  float mx = -INFINITY;
#pragma unroll
  for (int o = 0; o < 44; ++o) mx = fmaxf(mx, s[o]);
  float sum = 0.0f;
#pragma unroll
  for (int o = 0; o < 44; ++o) { float e = expf(s[o] - mx); s[o] = e; sum += e; }
  float inv = 1.0f / sum;

  float4 acc[16];
#pragma unroll
  for (int j = 0; j < 16; ++j) acc[j] = make_float4(0.f, 0.f, 0.f, 0.f);
#pragma unroll
  for (int o = 0; o < 44; ++o) {
    int d = c_off[o];
    if (i >= d) {
      float a = s[o] * inv;
      const float4* vp = (const float4*)(qkv + (size_t)(i - d) * 3072 + 2048 + h * 64);
#pragma unroll
      for (int j = 0; j < 16; ++j) {
        float4 vv = vp[j];
        acc[j].x += a * vv.x; acc[j].y += a * vv.y;
        acc[j].z += a * vv.z; acc[j].w += a * vv.w;
      }
    }
  }
  float4* op = (float4*)(out + (size_t)i * 1024 + h * 64);
#pragma unroll
  for (int j = 0; j < 16; ++j) op[j] = acc[j];
}

// ---------------- sigmoid gate fuse -> f16 ----------
__global__ __launch_bounds__(256) void gate_fuse_kernel(
    const float* __restrict__ attn, const float* __restrict__ glin,
    _Float16* __restrict__ out, int n) {
  int i = blockIdx.x * 256 + threadIdx.x;
  if (i < n) {
    float g = 1.0f / (1.0f + expf(-glin[i]));
    out[i] = (_Float16)(attn[i] * g);
  }
}

// ---------------- launch ----------------------------
extern "C" void kernel_launch(void* const* d_in, const int* in_sizes, int n_in,
                              void* d_out, int out_size, void* d_ws, size_t ws_size,
                              hipStream_t stream) {
  const int N = 2048, D = 1024, H = 16, FF = 4096, D3 = 3072;

  const float* x      = (const float*)d_in[0];
  const float* ln1_g  = (const float*)d_in[1];
  const float* ln1_b  = (const float*)d_in[2];
  const float* qkv_w  = (const float*)d_in[3];
  const float* qkv_b  = (const float*)d_in[4];
  const float* gate_w = (const float*)d_in[5];
  const float* gate_b = (const float*)d_in[6];
  const float* out_w  = (const float*)d_in[7];
  const float* out_b  = (const float*)d_in[8];
  const float* posb   = (const float*)d_in[9];
  const float* ln2_g  = (const float*)d_in[10];
  const float* ln2_b  = (const float*)d_in[11];
  const float* fc1_w  = (const float*)d_in[12];
  const float* fc1_b  = (const float*)d_in[13];
  const float* fc2_w  = (const float*)d_in[14];
  const float* fc2_b  = (const float*)d_in[15];
  float* y = (float*)d_out;

  char* p = (char*)d_ws;
  auto alloc = [&](size_t bytes) { void* r = (void*)p; p += (bytes + 255) & ~(size_t)255; return r; };
  _Float16* wqkv_t  = (_Float16*)alloc((size_t)D * D3 * 2);   // [3072][1024]
  _Float16* wgate_t = (_Float16*)alloc((size_t)D * D * 2);    // [1024][1024]
  _Float16* wout_t  = (_Float16*)alloc((size_t)D * D * 2);
  _Float16* wfc1_t  = (_Float16*)alloc((size_t)D * FF * 2);   // [4096][1024]
  _Float16* wfc2_t  = (_Float16*)alloc((size_t)FF * D * 2);   // [1024][4096]
  _Float16* xn_h    = (_Float16*)alloc((size_t)N * D * 2);    // reused for LN2
  float*    qkv_f   = (float*)   alloc((size_t)N * D3 * 4);
  float*    gate_f  = (float*)   alloc((size_t)N * D * 4);
  float*    attn_f  = (float*)   alloc((size_t)N * D * 4);
  _Float16* gated_h = (_Float16*)alloc((size_t)N * D * 2);
  float*    x1_f    = (float*)   alloc((size_t)N * D * 4);
  _Float16* ff1_h   = (_Float16*)alloc((size_t)N * FF * 2);

  dim3 blkT(32, 8);
  cast_transpose_kernel<<<dim3(D3 / 32, D / 32), blkT, 0, stream>>>(qkv_w,  wqkv_t,  D, D3);
  cast_transpose_kernel<<<dim3(D  / 32, D / 32), blkT, 0, stream>>>(gate_w, wgate_t, D, D);
  cast_transpose_kernel<<<dim3(D  / 32, D / 32), blkT, 0, stream>>>(out_w,  wout_t,  D, D);
  cast_transpose_kernel<<<dim3(FF / 32, D / 32), blkT, 0, stream>>>(fc1_w,  wfc1_t,  D, FF);
  cast_transpose_kernel<<<dim3(D / 32, FF / 32), blkT, 0, stream>>>(fc2_w,  wfc2_t,  FF, D);

  // LN1 -> xn (f16)
  ln_kernel<<<N, 256, 0, stream>>>(x, ln1_g, ln1_b, xn_h);

  // QKV + gate projections
  gemm_wmma<0><<<dim3(D3 / BN, N / BM), 256, 0, stream>>>(
      xn_h, wqkv_t, qkv_b, nullptr, qkv_f, nullptr, N, D3, D);
  gemm_wmma<0><<<dim3(D / BN, N / BM), 256, 0, stream>>>(
      xn_h, wgate_t, gate_b, nullptr, gate_f, nullptr, N, D, D);

  // offset attention
  attn_kernel<<<(N * H) / 256, 256, 0, stream>>>(qkv_f, posb, attn_f, N);

  // gate fuse -> f16
  gate_fuse_kernel<<<(N * D) / 256, 256, 0, stream>>>(attn_f, gate_f, gated_h, N * D);

  // out projection + residual -> x1
  gemm_wmma<1><<<dim3(D / BN, N / BM), 256, 0, stream>>>(
      gated_h, wout_t, out_b, x, x1_f, nullptr, N, D, D);

  // LN2 -> xn (reuse)
  ln_kernel<<<N, 256, 0, stream>>>(x1_f, ln2_g, ln2_b, xn_h);

  // fc1 + exact GELU -> f16
  gemm_wmma<2><<<dim3(FF / BN, N / BM), 256, 0, stream>>>(
      xn_h, wfc1_t, fc1_b, nullptr, nullptr, ff1_h, N, FF, D);

  // fc2 + residual -> output
  gemm_wmma<1><<<dim3(D / BN, N / BM), 256, 0, stream>>>(
      ff1_h, wfc2_t, fc2_b, x1_f, y, nullptr, N, D, FF);
}
